// ESN_12773232738532
// MI455X (gfx1250) — matrix operand order repaired and checked by reading, not compiled, not run on hardware
//
#include <hip/hip_runtime.h>
#include <math.h>

// ---------------------------------------------------------------------------
// ESN on gfx1250: persistent bf16-WMMA recurrence with TDM (tensor_load_to_lds)
// staging of the hidden state + LDS K-split reduction, and WMMA readout GEMMs.
//   recurrence: h = 0.5h + 0.5*tanh(x_t@W_in^T + h@W_hat^T), T=1024 serial
//   readout:    GELU(GELU(hs@W0^T+b0)@W1^T+b1)@W2^T+b2
// ---------------------------------------------------------------------------

typedef __attribute__((ext_vector_type(16))) __bf16 v16bf;
typedef __attribute__((ext_vector_type(8)))  __bf16 v8bf;
typedef __attribute__((ext_vector_type(8)))  float  v8f;
typedef __attribute__((ext_vector_type(4)))  unsigned int u32x4;
typedef __attribute__((ext_vector_type(8)))  int    i32x8;
typedef __attribute__((ext_vector_type(4)))  int    i32x4;

#define RES_  2048
#define IN_   64
#define OUT_  64
#define B_    32
#define T_    1024
#define H1_   512

// ---- WMMA fragment loaders from row-major bf16 ---------------------------
// A (16x32, MxK), lane<16: M=lane, K={k0..k0+7, k0+16..k0+23}
//                lane>=16: M=lane-16, K={k0+8..k0+15, k0+24..k0+31}
__device__ __forceinline__ v16bf load_a_frag(const __bf16* A, int ld, int m0,
                                             int k0, int lane) {
  int m   = m0 + (lane & 15);
  int sel = lane >> 4;
  const __bf16* p = A + (size_t)m * ld + k0 + sel * 8;
  v8bf lo = *(const v8bf*)(p);        // K = k0 + sel*8 + 0..7
  v8bf hi = *(const v8bf*)(p + 16);   // K = k0 + 16 + sel*8 + 0..7
  return __builtin_shufflevector(lo, hi, 0, 1, 2, 3, 4, 5, 6, 7,
                                 8, 9, 10, 11, 12, 13, 14, 15);
}

// B (32x16, KxN) with B[k][n] = W[n][k], W row-major (N x K):
// lane holds column n = lane&15, K = {k0+(lane>>4)*8+0..7, k0+16+(lane>>4)*8+0..7}.
__device__ __forceinline__ v16bf load_b_frag(const __bf16* W, int ld, int n0,
                                             int k0, int lane) {
  int n = n0 + (lane & 15);
  int k = k0 + (lane >> 4) * 8;
  const __bf16* p = W + (size_t)n * ld + k;
  v8bf lo = *(const v8bf*)(p);
  v8bf hi = *(const v8bf*)(p + 16);
  return __builtin_shufflevector(lo, hi, 0, 1, 2, 3, 4, 5, 6, 7,
                                 8, 9, 10, 11, 12, 13, 14, 15);
}

#define WMMA_BF16(a, b, c) \
  __builtin_amdgcn_wmma_f32_16x16x32_bf16(false, (a), false, (b), (short)0, \
                                          (c), false, false)

// ---- Tensor Data Mover: 2D tile (global, row-major) -> LDS ---------------
// data_size = 2 bytes (bf16). Descriptor per CDNA5 ISA ch.10.8 / D# groups.
__device__ __forceinline__ void tdm_load_2d_bf16(unsigned lds_addr,
                                                 const void* gptr,
                                                 unsigned dim0, unsigned dim1,
                                                 unsigned stride0) {
  unsigned long long ga = (unsigned long long)(size_t)gptr;
  u32x4 g0;
  g0[0] = 1u;                                   // count=1, user descriptor
  g0[1] = lds_addr;                             // LDS byte address
  g0[2] = (unsigned)(ga & 0xffffffffull);       // global_addr[31:0]
  g0[3] = (unsigned)((ga >> 32) & 0x01ffffffull) | (2u << 30);  // type=2
  i32x8 g1;
  g1[0] = (int)(1u << 16);                      // data_size=1 (2 bytes)
  g1[1] = (int)((dim0 & 0xffffu) << 16);        // tensor_dim0[15:0]
  g1[2] = (int)(((dim0 >> 16) & 0xffffu) | ((dim1 & 0xffffu) << 16));
  g1[3] = (int)(((dim1 >> 16) & 0xffffu) | ((dim0 & 0xffffu) << 16));  // tile_dim0
  g1[4] = (int)(dim1 & 0xffffu);                // tile_dim1 (tile == tensor)
  g1[5] = (int)stride0;                         // tensor_dim0_stride[31:0]
  g1[6] = 0;
  g1[7] = 0;
  i32x4 z4 = {0, 0, 0, 0};
#if defined(__clang_major__) && __clang_major__ >= 23
  i32x8 z8 = {0, 0, 0, 0, 0, 0, 0, 0};
  __builtin_amdgcn_tensor_load_to_lds(g0, g1, z4, z4, z8, 0);
#else
  __builtin_amdgcn_tensor_load_to_lds(g0, g1, z4, z4, 0);
#endif
}

// ---- fp32 -> bf16 conversion ---------------------------------------------
__global__ void f32_to_bf16(const float* __restrict__ in,
                            __bf16* __restrict__ out, int n) {
  for (int i = blockIdx.x * blockDim.x + threadIdx.x; i < n;
       i += gridDim.x * blockDim.x)
    out[i] = (__bf16)in[i];
}

// ---- device-wide sense-reversing barrier ---------------------------------
__device__ __forceinline__ void grid_sync(unsigned* bar, unsigned nblocks) {
  __syncthreads();
  if (threadIdx.x == 0) {
    __threadfence();
    unsigned gen =
        __hip_atomic_load(bar + 1, __ATOMIC_RELAXED, __HIP_MEMORY_SCOPE_AGENT);
    if (__hip_atomic_fetch_add(bar, 1u, __ATOMIC_ACQ_REL,
                               __HIP_MEMORY_SCOPE_AGENT) == nblocks - 1) {
      __hip_atomic_store(bar, 0u, __ATOMIC_RELAXED, __HIP_MEMORY_SCOPE_AGENT);
      __hip_atomic_fetch_add(bar + 1, 1u, __ATOMIC_RELEASE,
                             __HIP_MEMORY_SCOPE_AGENT);
    } else {
      while (__hip_atomic_load(bar + 1, __ATOMIC_ACQUIRE,
                               __HIP_MEMORY_SCOPE_AGENT) == gen)
        __builtin_amdgcn_s_sleep(2);
    }
  }
  __syncthreads();
}

// ---- persistent recurrence kernel ----------------------------------------
// 32 blocks x 8 waves. Per block: 4 N-tiles x 2 K-slices. Each wave computes
// a full 32x16 tile (both M-tiles, shared B loads) over half of K, then the
// wave pair reduces partial sums through LDS. h is staged into LDS once per
// block per step by the Tensor Data Mover.
extern __shared__ char dynsmem[];

__global__ void __launch_bounds__(256) esn_recur(
    const __bf16* __restrict__ xb,     // (B,T,IN) bf16
    const __bf16* __restrict__ Winb,   // (RES,IN) bf16
    const __bf16* __restrict__ Whatb,  // (RES,RES) bf16
    float* __restrict__ hf0, float* __restrict__ hf1,    // fp32 state pingpong
    __bf16* __restrict__ hb0, __bf16* __restrict__ hb1,  // bf16 state pingpong
    __bf16* __restrict__ hs,           // (B,T,RES) bf16 trajectory
    unsigned* __restrict__ bar) {
  const int lane  = threadIdx.x & 31;
  const int w     = threadIdx.x >> 5;        // 0..7
  const int ks    = w & 1;                   // K slice
  const int ntile = blockIdx.x * 4 + (w >> 1);
  const int n     = ntile * 16 + (lane & 15);
  const int sel   = lane >> 4;

  __bf16* sA  = (__bf16*)dynsmem;                          // 32x2048 bf16
  float*  red = (float*)(dynsmem + (size_t)B_ * RES_ * 2); // 8KB reduction
  const unsigned ldsA = (unsigned)(size_t)sA;              // LDS byte addr

  for (int t = 0; t < T_; ++t) {
    const __bf16* hprev_b = (t & 1) ? hb1 : hb0;
    const float*  hprev_f = (t & 1) ? hf1 : hf0;
    float*        hnext_f = (t & 1) ? hf0 : hf1;
    __bf16*       hnext_b = (t & 1) ? hb0 : hb1;

    // --- stage h_t (32x2048 bf16, 128KB) into LDS via TDM (wave 0 issues)
    if (w == 0) {
      tdm_load_2d_bf16(ldsA, hprev_b, RES_, B_, RES_);
      __builtin_amdgcn_s_wait_tensorcnt(0);
    }
    __syncthreads();

    v8f acc0 = {}, acc1 = {};
    // --- input drive: K=64 split between the wave pair (one 32-chunk each)
    {
      int k0 = ks * 32;
      v16bf a0 = load_a_frag(xb + t * IN_, T_ * IN_, 0, k0, lane);
      v16bf a1 = load_a_frag(xb + t * IN_, T_ * IN_, 16, k0, lane);
      v16bf b  = load_b_frag(Winb, IN_, ntile * 16, k0, lane);
      acc0 = WMMA_BF16(a0, b, acc0);
      acc1 = WMMA_BF16(a1, b, acc1);
    }
    // --- recurrent drive: this wave's half of K (A from LDS, B from L2)
#pragma unroll 4
    for (int k0 = ks * (RES_ / 2); k0 < (ks + 1) * (RES_ / 2); k0 += 32) {
      v16bf a0 = load_a_frag(sA, RES_, 0, k0, lane);
      v16bf a1 = load_a_frag(sA, RES_, 16, k0, lane);
      v16bf b  = load_b_frag(Whatb, RES_, ntile * 16, k0, lane);
      acc0 = WMMA_BF16(a0, b, acc0);
      acc1 = WMMA_BF16(a1, b, acc1);
    }
    // --- K-split reduction through LDS: wave ks finalizes M-tile == ks
    *(v8f*)(red + (w * 32 + lane) * 8) = ks ? acc0 : acc1;
    __syncthreads();
    v8f mine = ks ? acc1 : acc0;
    mine += *(const v8f*)(red + ((w ^ 1) * 32 + lane) * 8);

    // --- epilogue: leak + tanh; rows m = ks*16 .. ks*16+15
#pragma unroll
    for (int j = 0; j < 8; ++j) {
      int m = ks * 16 + sel * 8 + j;  // batch index
      float hp = hprev_f[m * RES_ + n];
      float hn = 0.5f * hp + 0.5f * tanhf(mine[j]);
      hnext_f[m * RES_ + n] = hn;
      hnext_b[m * RES_ + n] = (__bf16)hn;
      hs[((size_t)m * T_ + t) * RES_ + n] = (__bf16)hn;
    }
    grid_sync(bar, gridDim.x);
  }
}

// ---- generic readout GEMM: out = act(A @ W^T + bias) ---------------------
template <int ACT /*0 none, 1 exact gelu*/, int OUT_BF16>
__global__ void __launch_bounds__(256) gemm_bias_act(
    const __bf16* __restrict__ A, const __bf16* __restrict__ W,
    const float* __restrict__ bias, void* __restrict__ out, int M, int N,
    int K) {
  const int lane = threadIdx.x & 31;
  const int wid  = blockIdx.x * 8 + (threadIdx.x >> 5);
  const int ntiles = N >> 4;
  const int mtile  = wid / ntiles;
  const int ntile  = wid % ntiles;
  if (mtile * 16 >= M) return;  // wave-uniform

  v8f acc = {};
#pragma unroll 4
  for (int k0 = 0; k0 < K; k0 += 32) {
    v16bf a = load_a_frag(A, K, mtile * 16, k0, lane);
    v16bf b = load_b_frag(W, K, ntile * 16, k0, lane);
    acc = WMMA_BF16(a, b, acc);
  }
  const int n = ntile * 16 + (lane & 15);
  const float bn = bias[n];
#pragma unroll
  for (int j = 0; j < 8; ++j) {
    int m = mtile * 16 + (lane >> 4) * 8 + j;
    float v = acc[j] + bn;
    if (ACT == 1) v = 0.5f * v * (1.0f + erff(v * 0.70710678118654752f));
    if (OUT_BF16)
      ((__bf16*)out)[(size_t)m * N + n] = (__bf16)v;
    else
      ((float*)out)[(size_t)m * N + n] = v;
  }
}

// ---------------------------------------------------------------------------
extern "C" void kernel_launch(void* const* d_in, const int* in_sizes, int n_in,
                              void* d_out, int out_size, void* d_ws,
                              size_t ws_size, hipStream_t stream) {
  const float* x     = (const float*)d_in[0];  // (B,T,IN)
  const float* W_in  = (const float*)d_in[1];  // (RES,IN)
  const float* W_hat = (const float*)d_in[2];  // (RES,RES)
  const float* W0    = (const float*)d_in[3];  // (512,RES)
  const float* b0    = (const float*)d_in[4];
  const float* W1    = (const float*)d_in[5];  // (512,512)
  const float* b1    = (const float*)d_in[6];
  const float* W2    = (const float*)d_in[7];  // (64,512)
  const float* b2    = (const float*)d_in[8];

  char* p = (char*)d_ws;
  auto carve = [&](size_t bytes) {
    char* r = p;
    p += (bytes + 255) & ~(size_t)255;
    return r;
  };
  __bf16* xb    = (__bf16*)carve((size_t)B_ * T_ * IN_ * 2);
  __bf16* Winb  = (__bf16*)carve((size_t)RES_ * IN_ * 2);
  __bf16* Whatb = (__bf16*)carve((size_t)RES_ * RES_ * 2);
  __bf16* W0b   = (__bf16*)carve((size_t)H1_ * RES_ * 2);
  __bf16* W1b   = (__bf16*)carve((size_t)H1_ * H1_ * 2);
  __bf16* W2b   = (__bf16*)carve((size_t)OUT_ * H1_ * 2);
  float*  hf0   = (float*)carve((size_t)B_ * RES_ * 4);
  float*  hf1   = (float*)carve((size_t)B_ * RES_ * 4);
  __bf16* hb0   = (__bf16*)carve((size_t)B_ * RES_ * 2);
  __bf16* hb1   = (__bf16*)carve((size_t)B_ * RES_ * 2);
  __bf16* hs    = (__bf16*)carve((size_t)B_ * T_ * RES_ * 2);   // 128 MB
  __bf16* y1    = (__bf16*)carve((size_t)B_ * T_ * H1_ * 2);    // 32 MB
  __bf16* y2    = (__bf16*)carve((size_t)B_ * T_ * H1_ * 2);    // 32 MB
  unsigned* bar = (unsigned*)carve(256);

  // reset barrier + zero initial state (t=0 reads buffer 0)
  hipMemsetAsync(bar, 0, 256, stream);
  hipMemsetAsync(hf0, 0, (size_t)B_ * RES_ * 4, stream);
  hipMemsetAsync(hb0, 0, (size_t)B_ * RES_ * 2, stream);

  // fp32 -> bf16 weight / input conversion
  f32_to_bf16<<<1024, 256, 0, stream>>>(x, xb, B_ * T_ * IN_);
  f32_to_bf16<<<512, 256, 0, stream>>>(W_in, Winb, RES_ * IN_);
  f32_to_bf16<<<2048, 256, 0, stream>>>(W_hat, Whatb, RES_ * RES_);
  f32_to_bf16<<<1024, 256, 0, stream>>>(W0, W0b, H1_ * RES_);
  f32_to_bf16<<<512, 256, 0, stream>>>(W1, W1b, H1_ * H1_);
  f32_to_bf16<<<128, 256, 0, stream>>>(W2, W2b, OUT_ * H1_);

  // serial recurrence: 1024 steps inside one persistent kernel.
  // Dynamic LDS: 128KB staged h + 8KB reduction scratch.
  const size_t smem = (size_t)B_ * RES_ * 2 + 8 * 32 * 8 * sizeof(float);
  esn_recur<<<32, 256, smem, stream>>>(xb, Winb, Whatb, hf0, hf1, hb0, hb1, hs,
                                       bar);

  // h_n output: after t=1023 the final state lives in buffer 0
  hipMemcpyAsync((float*)d_out + (size_t)B_ * T_ * OUT_, hf0,
                 (size_t)B_ * RES_ * 4, hipMemcpyDeviceToDevice, stream);

  // readout: (32768 x 2048) -> 512 -> 512 -> 64
  const int M = B_ * T_;
  gemm_bias_act<1, 1><<<(M / 16) * (H1_ / 16) / 8, 256, 0, stream>>>(
      hs, W0b, b0, y1, M, H1_, RES_);
  gemm_bias_act<1, 1><<<(M / 16) * (H1_ / 16) / 8, 256, 0, stream>>>(
      y1, W1b, b1, y2, M, H1_, H1_);
  gemm_bias_act<0, 0><<<(M / 16) * (OUT_ / 16) / 8, 256, 0, stream>>>(
      y2, W2b, b2, d_out, M, OUT_, H1_);
}
